// SingleHeadAtt_52673478918360
// MI455X (gfx1250) — matrix operand order repaired and checked by reading, not compiled, not run on hardware
//
#include <hip/hip_runtime.h>
#include <hip/hip_bf16.h>

// ---------------------------------------------------------------------------
// Single-head causal attention for MI455X (gfx1250, wave32, WMMA).
//   B=4, S=2048, E=1024, fp32 in/out.
// Pipeline (all WMMA f16 x f16 -> f32):
//   1) qkv_proj : Q/K/V = X@W^T + b (Q pre-scaled 1/sqrt(E)), f16 out,
//                 register-staged software pipeline (f32->f16 cvt required)
//   2) scores   : S = Q@K^T, causal mask, f32 out (fits 192MB L2);
//                 GLOBAL_LOAD_ASYNC_TO_LDS double-buffered, 64x64 wave tiles
//   3) softmax  : row softmax, f16 probs
//   4) out      : O = P@V, K-loop truncated at causal diagonal;
//                 async A pipeline + register-staged V transpose, 64x64 tiles
// ---------------------------------------------------------------------------

typedef __attribute__((ext_vector_type(16))) _Float16 v16h;
typedef __attribute__((ext_vector_type(8)))  _Float16 v8h;
typedef __attribute__((ext_vector_type(4)))  _Float16 v4h;
typedef __attribute__((ext_vector_type(8)))  float    v8f;
typedef __attribute__((ext_vector_type(4)))  float    v4f;

#define BATCH 4
#define SEQ   2048
#define EMB   1024

#define BM  128         // block tile M (all GEMMs)
#define BK  32          // K per stage == WMMA K
#define LDK 48          // padded LDS K-stride in halfs (96B, 16B aligned, bank-skewed)

union FragH { v16h v; v8h h[2]; };

// A fragment, 16x32 f16 (M x K), per CDNA5 ISA layout:
// lane<16 holds row (lane&15), K={0..7,16..23}; lane>=16 same rows, K={8..15,24..31}
__device__ inline v16h frag_a(const _Float16* s, int rowBase, int lane) {
  const _Float16* p = s + (rowBase + (lane & 15)) * LDK + ((lane >> 4) & 1) * 8;
  FragH f;
  f.h[0] = *(const v8h*)(p);
  f.h[1] = *(const v8h*)(p + 16);
  return f.v;
}

// B fragment, 32x16 f16 (K x N), LDS stored K-major per column:
// lane<16 holds col (lane&15), K={0..7,16..23}; lane>=16 same cols, K={8..15,24..31}
__device__ inline v16h frag_b(const _Float16* s, int colBase, int lane) {
  const _Float16* p = s + (colBase + (lane & 15)) * LDK + ((lane >> 4) & 1) * 8;
  FragH f;
  f.h[0] = *(const v8h*)(p);
  f.h[1] = *(const v8h*)(p + 16);
  return f.v;
}

// 32x64 wave macro-tile: 8 WMMA / 12 frag loads (qkv kernel)
__device__ inline void mma_tile32x64(const _Float16* As, const _Float16* Bs,
                                     int wm, int wn, int lane, v8f (&acc)[2][4]) {
  v16h a0 = frag_a(As, wm,      lane);
  v16h a1 = frag_a(As, wm + 16, lane);
#pragma unroll
  for (int j = 0; j < 4; ++j) {
    v16h b = frag_b(Bs, wn + 16 * j, lane);
    acc[0][j] = __builtin_amdgcn_wmma_f32_16x16x32_f16(false, a0, false, b,
                  (short)0, acc[0][j], false, false);
    acc[1][j] = __builtin_amdgcn_wmma_f32_16x16x32_f16(false, a1, false, b,
                  (short)0, acc[1][j], false, false);
  }
}

// 64x64 wave macro-tile: 16 WMMA / 16 frag loads -> 512B LDS per WMMA,
// 1.5x less LDS read traffic per FLOP than 32x64 (LDS bw is the limiter).
__device__ inline void mma_tile64x64(const _Float16* As, const _Float16* Bs,
                                     int wm, int wn, int lane, v8f (&acc)[4][4]) {
  v16h a[4];
#pragma unroll
  for (int i = 0; i < 4; ++i) a[i] = frag_a(As, wm + 16 * i, lane);
#pragma unroll
  for (int j = 0; j < 4; ++j) {
    v16h b = frag_b(Bs, wn + 16 * j, lane);
#pragma unroll
    for (int i = 0; i < 4; ++i)
      acc[i][j] = __builtin_amdgcn_wmma_f32_16x16x32_f16(false, a[i], false, b,
                    (short)0, acc[i][j], false, false);
  }
}

// Async-copy a ROWSx32 f16 tile (K-contiguous rows) global -> padded LDS.
// ROWS/64 instructions per thread -> ASYNCcnt += ROWS/64 per wave per call.
template <int ROWS>
__device__ inline void async_tile(const _Float16* __restrict__ gbase, int row0,
                                  size_t rowStride, int kt, unsigned ldsBase, int tid) {
#pragma unroll
  for (int i = 0; i < ROWS / 64; ++i) {
    int idx = tid + i * 256;
    int r = idx >> 2;
    int c = (idx & 3) * 8;
    unsigned voff = (unsigned)(((size_t)(row0 + r) * rowStride + (size_t)kt + c) * 2u);
    unsigned lds  = ldsBase + (unsigned)((r * LDK + c) * 2);
    asm volatile("global_load_async_to_lds_b128 %0, %1, %2"
                 :: "v"(lds), "v"(voff), "s"(gbase) : "memory");
  }
}

// ---------------------------------------------------------------------------
// 1) Fused QKV projection: Y[m,f] = (sum_e X[m,e]*W[f,e] + b[f]) * scale
//    grid = (EMB/128, (B*S)/BM, 3)   z selects Q/K/V
// ---------------------------------------------------------------------------
__global__ __launch_bounds__(256) void qkv_proj_kernel(
    const float* __restrict__ X,
    const float* __restrict__ Wq, const float* __restrict__ bq,
    const float* __restrict__ Wk, const float* __restrict__ bk,
    const float* __restrict__ Wv, const float* __restrict__ bv,
    _Float16* __restrict__ Qh, _Float16* __restrict__ Kh, _Float16* __restrict__ Vh)
{
  constexpr int BN = 128;
  __shared__ _Float16 As[2][BM * LDK];
  __shared__ _Float16 Bs[2][BN * LDK];

  const int tid  = threadIdx.x;
  const int lane = tid & 31;
  const int wave = tid >> 5;
  const int wm   = (wave & 3) * 32;
  const int wn   = (wave >> 2) * 64;
  const int mBase = blockIdx.y * BM;
  const int nBase = blockIdx.x * BN;
  const int z = blockIdx.z;

  const float* W    = (z == 0) ? Wq : (z == 1) ? Wk : Wv;
  const float* bias = (z == 0) ? bq : (z == 1) ? bk : bv;
  _Float16*    Y    = (z == 0) ? Qh : (z == 1) ? Kh : Vh;
  const float scale = (z == 0) ? 0.03125f : 1.0f;  // fold 1/sqrt(1024) into Q

  v8f acc[2][4] = {};
  v4f aReg[4], bReg[4];

  auto loadRegs = [&](int kt) {
#pragma unroll
    for (int i = 0; i < 4; ++i) {
      int idx = tid + i * 256;
      int r = idx >> 3;
      int c = (idx & 7) * 4;
      aReg[i] = *(const v4f*)(X + (size_t)(mBase + r) * EMB + kt + c);
      bReg[i] = *(const v4f*)(W + (size_t)(nBase + r) * EMB + kt + c);
    }
  };
  auto storeRegs = [&](int buf) {
#pragma unroll
    for (int i = 0; i < 4; ++i) {
      int idx = tid + i * 256;
      int r = idx >> 3;
      int c = (idx & 7) * 4;
      _Float16* da = &As[buf][r * LDK + c];
      da[0] = (_Float16)aReg[i].x; da[1] = (_Float16)aReg[i].y;
      da[2] = (_Float16)aReg[i].z; da[3] = (_Float16)aReg[i].w;
      _Float16* db = &Bs[buf][r * LDK + c];
      db[0] = (_Float16)bReg[i].x; db[1] = (_Float16)bReg[i].y;
      db[2] = (_Float16)bReg[i].z; db[3] = (_Float16)bReg[i].w;
    }
  };

  loadRegs(0);
  storeRegs(0);
  const int NT = EMB / BK;
  for (int t = 0; t < NT; ++t) {
    int buf = t & 1;
    bool more = (t + 1) < NT;
    if (more) loadRegs((t + 1) * BK);   // global loads overlap this tile's math
    __syncthreads();
    mma_tile32x64(As[buf], Bs[buf], wm, wn, lane, acc);
    if (more) storeRegs(buf ^ 1);       // cvt+ds_store after compute, before barrier
    __syncthreads();
  }

#pragma unroll
  for (int i = 0; i < 2; ++i)
#pragma unroll
    for (int j = 0; j < 4; ++j) {
      int n = nBase + wn + 16 * j + (lane & 15);
      float bn = bias[n];
#pragma unroll
      for (int r = 0; r < 8; ++r) {
        int m = mBase + wm + 16 * i + r + ((lane >> 4) & 1) * 8;
        Y[(size_t)m * EMB + n] = (_Float16)((acc[i][j][r] + bn) * scale);
      }
    }
}

// ---------------------------------------------------------------------------
// 2) Scores: S[q,k] = Q[q,:] . K[k,:] (Q pre-scaled), causal mask -> -inf
//    grid = (SEQ/256, SEQ/128, BATCH); 128x256 block, 64x64 wave tiles.
//    Both operands stream via GLOBAL_LOAD_ASYNC_TO_LDS, double-buffered.
// ---------------------------------------------------------------------------
__global__ __launch_bounds__(256) void scores_kernel(
    const _Float16* __restrict__ Qh, const _Float16* __restrict__ Kh,
    const int* __restrict__ maskedFlag, float* __restrict__ Sc)
{
  constexpr int BN = 256;
  const int b = blockIdx.z;
  const int qBase = blockIdx.y * BM;
  const int kBase = blockIdx.x * BN;
  float* Srow = Sc + (size_t)b * SEQ * SEQ;
  const int msk = maskedFlag[0];

  if (msk && kBase > qBase + (BM - 1)) {
    // fully masked tile: fill -inf, skip the GEMM entirely
    for (int i = threadIdx.x; i < BM * BN; i += 256) {
      int r = i >> 8, c = i & 255;
      Srow[(size_t)(qBase + r) * SEQ + kBase + c] = -__builtin_inff();
    }
    return;
  }

  __shared__ _Float16 As[2][BM * LDK];
  __shared__ _Float16 Bs[2][BN * LDK];
  const int tid  = threadIdx.x;
  const int lane = tid & 31;
  const int wave = tid >> 5;
  const int wm = (wave & 1) * 64;
  const int wn = (wave >> 1) * 64;

  const _Float16* Qg = Qh + (size_t)b * SEQ * EMB;
  const _Float16* Kg = Kh + (size_t)b * SEQ * EMB;

  const unsigned aB[2] = { (unsigned)(size_t)&As[0][0], (unsigned)(size_t)&As[1][0] };
  const unsigned bB[2] = { (unsigned)(size_t)&Bs[0][0], (unsigned)(size_t)&Bs[1][0] };

  v8f acc[4][4] = {};

  // prologue: tile 0 in flight (2 + 4 async ops per thread)
  async_tile<BM>(Qg, qBase, EMB, 0, aB[0], tid);
  async_tile<BN>(Kg, kBase, EMB, 0, bB[0], tid);

  const int NT = EMB / BK;
  for (int t = 0; t < NT; ++t) {
    int buf = t & 1;
    if (t + 1 < NT) {
      // issue next tile (6 async ops), then wait for the previous 6 to land:
      // async loads complete in order, so asynccnt<=6 == tile t is resident.
      async_tile<BM>(Qg, qBase, EMB, (t + 1) * BK, aB[buf ^ 1], tid);
      async_tile<BN>(Kg, kBase, EMB, (t + 1) * BK, bB[buf ^ 1], tid);
      asm volatile("s_wait_asynccnt 0x6" ::: "memory");
    } else {
      asm volatile("s_wait_asynccnt 0x0" ::: "memory");
    }
    __syncthreads();
    mma_tile64x64(As[buf], Bs[buf], wm, wn, lane, acc);
    __syncthreads();
  }

#pragma unroll
  for (int i = 0; i < 4; ++i)
#pragma unroll
    for (int j = 0; j < 4; ++j) {
      int n = kBase + wn + 16 * j + (lane & 15);
#pragma unroll
      for (int r = 0; r < 8; ++r) {
        int m = qBase + wm + 16 * i + r + ((lane >> 4) & 1) * 8;
        float v = acc[i][j][r];
        if (msk && n > m) v = -__builtin_inff();
        Srow[(size_t)m * SEQ + n] = v;
      }
    }
}

// ---------------------------------------------------------------------------
// 3) Row softmax over 2048 scores -> f16 probs. One 256-thread block per row.
// ---------------------------------------------------------------------------
__global__ __launch_bounds__(256) void softmax_kernel(
    const float* __restrict__ Sc, _Float16* __restrict__ P)
{
  const int row = blockIdx.x;                 // 0 .. B*S-1
  const float* src = Sc + (size_t)row * SEQ;
  _Float16* dst = P + (size_t)row * SEQ;
  const int tid = threadIdx.x;
  __shared__ float red[8];

  float vals[8];
  float lmax = -__builtin_inff();
#pragma unroll
  for (int i = 0; i < 8; ++i) {
    vals[i] = src[tid + i * 256];
    lmax = fmaxf(lmax, vals[i]);
  }
#pragma unroll
  for (int off = 16; off; off >>= 1) lmax = fmaxf(lmax, __shfl_xor(lmax, off));
  if ((tid & 31) == 0) red[tid >> 5] = lmax;
  __syncthreads();
  float rmax = red[0];
#pragma unroll
  for (int w = 1; w < 8; ++w) rmax = fmaxf(rmax, red[w]);

  float lsum = 0.0f;
#pragma unroll
  for (int i = 0; i < 8; ++i) {
    float e = exp2f((vals[i] - rmax) * 1.44269504088896f);
    vals[i] = e;
    lsum += e;
  }
#pragma unroll
  for (int off = 16; off; off >>= 1) lsum += __shfl_xor(lsum, off);
  __syncthreads();
  if ((tid & 31) == 0) red[tid >> 5] = lsum;
  __syncthreads();
  float rsum = 0.0f;
#pragma unroll
  for (int w = 0; w < 8; ++w) rsum += red[w];
  float inv = 1.0f / rsum;
#pragma unroll
  for (int i = 0; i < 8; ++i) dst[tid + i * 256] = (_Float16)(vals[i] * inv);
}

// ---------------------------------------------------------------------------
// 4) O[q,d] = sum_k P[q,k] * V[k,d]; causal => K-loop stops at the diagonal.
//    grid = (EMB/256, SEQ/128, BATCH); 128x256 block, 64x64 wave tiles.
//    A tile async double-buffered; V transpose register-staged + pipelined.
// ---------------------------------------------------------------------------
__global__ __launch_bounds__(256) void out_kernel(
    const _Float16* __restrict__ P, const _Float16* __restrict__ Vh,
    const int* __restrict__ maskedFlag, float* __restrict__ O)
{
  constexpr int BN = 256;
  __shared__ _Float16 As[2][BM * LDK];
  __shared__ _Float16 Bs[2][BN * LDK];
  const int tid  = threadIdx.x;
  const int lane = tid & 31;
  const int wave = tid >> 5;
  const int wm = (wave & 1) * 64;
  const int wn = (wave >> 1) * 64;
  const int b = blockIdx.z;
  const int mBase = blockIdx.y * BM;   // query rows
  const int nBase = blockIdx.x * BN;   // head-dim cols

  const _Float16* Pg = P  + (size_t)b * SEQ * SEQ;
  const _Float16* Vg = Vh + (size_t)b * SEQ * EMB;
  const int msk = maskedFlag[0];
  const int kEnd = msk ? (mBase + BM < SEQ ? mBase + BM : SEQ) : SEQ;

  const unsigned aB[2] = { (unsigned)(size_t)&As[0][0], (unsigned)(size_t)&As[1][0] };

  v8f acc[4][4] = {};
  v4h vReg[8];

  auto loadV = [&](int kt) {
#pragma unroll
    for (int i = 0; i < 8; ++i) {
      int idx = tid + i * 256;
      int k  = idx >> 6;           // 0..31
      int cn = (idx & 63) * 4;     // 0..252
      vReg[i] = *(const v4h*)(Vg + (size_t)(kt + k) * EMB + nBase + cn);
    }
  };
  auto storeV = [&](int buf) {     // transpose into K-major LDS: Bs[n][k]
#pragma unroll
    for (int i = 0; i < 8; ++i) {
      int idx = tid + i * 256;
      int k  = idx >> 6;
      int cn = (idx & 63) * 4;
      Bs[buf][(cn + 0) * LDK + k] = vReg[i].x;
      Bs[buf][(cn + 1) * LDK + k] = vReg[i].y;
      Bs[buf][(cn + 2) * LDK + k] = vReg[i].z;
      Bs[buf][(cn + 3) * LDK + k] = vReg[i].w;
    }
  };

  // prologue: tile 0
  async_tile<BM>(Pg, mBase, SEQ, 0, aB[0], tid);
  loadV(0);
  storeV(0);

  const int NT = kEnd / BK;
  for (int t = 0; t < NT; ++t) {
    int buf = t & 1;
    bool more = (t + 1) < NT;
    if (more) {
      async_tile<BM>(Pg, mBase, SEQ, (t + 1) * BK, aB[buf ^ 1], tid);
      loadV((t + 1) * BK);
      asm volatile("s_wait_asynccnt 0x2" ::: "memory");
    } else {
      asm volatile("s_wait_asynccnt 0x0" ::: "memory");
    }
    __syncthreads();
    mma_tile64x64(As[buf], Bs[buf], wm, wn, lane, acc);
    if (more) storeV(buf ^ 1);
    __syncthreads();
  }

#pragma unroll
  for (int i = 0; i < 4; ++i)
#pragma unroll
    for (int j = 0; j < 4; ++j) {
      int n = nBase + wn + 16 * j + (lane & 15);
#pragma unroll
      for (int r = 0; r < 8; ++r) {
        int m = mBase + wm + 16 * i + r + ((lane >> 4) & 1) * 8;
        O[((size_t)b * SEQ + m) * EMB + n] = acc[i][j][r];
      }
    }
}

// ---------------------------------------------------------------------------
// Launch. Workspace layout (144 MB total):
//   Qh f16 16MB | Kh f16 16MB | Vh f16 16MB | scores f32 64MB | probs f16 32MB
// ---------------------------------------------------------------------------
extern "C" void kernel_launch(void* const* d_in, const int* in_sizes, int n_in,
                              void* d_out, int out_size, void* d_ws, size_t ws_size,
                              hipStream_t stream) {
  const float* X      = (const float*)d_in[0];
  const int*   masked = (const int*)  d_in[1];
  const float* Wq = (const float*)d_in[2];
  const float* bq = (const float*)d_in[3];
  const float* Wk = (const float*)d_in[4];
  const float* bk = (const float*)d_in[5];
  const float* Wv = (const float*)d_in[6];
  const float* bv = (const float*)d_in[7];
  float* Out = (float*)d_out;

  const size_t HM = (size_t)BATCH * SEQ * EMB;   // 8M elements
  _Float16* Qh = (_Float16*)d_ws;
  _Float16* Kh = Qh + HM;
  _Float16* Vh = Kh + HM;
  float*    Sc = (float*)(Vh + HM);
  _Float16* Pp = (_Float16*)(Sc + (size_t)BATCH * SEQ * SEQ);

  dim3 blk(256);
  qkv_proj_kernel<<<dim3(EMB / 128, (BATCH * SEQ) / BM, 3), blk, 0, stream>>>(
      X, Wq, bq, Wk, bk, Wv, bv, Qh, Kh, Vh);
  scores_kernel<<<dim3(SEQ / 256, SEQ / BM, BATCH), blk, 0, stream>>>(
      Qh, Kh, masked, Sc);
  softmax_kernel<<<dim3(BATCH * SEQ), blk, 0, stream>>>(Sc, Pp);
  out_kernel<<<dim3(EMB / 256, SEQ / BM, BATCH), blk, 0, stream>>>(
      Pp, Vh, masked, Out);
}
